// FeedBack_41678362640751
// MI455X (gfx1250) — matrix operand order repaired and checked by reading, not compiled, not run on hardware
//
#include <hip/hip_runtime.h>
#include <hip/hip_bf16.h>

// ---------------------------------------------------------------------------
// 2-layer GRU (UNITS=256), batch 4096, 191 sequential steps.
// Persistent blocks, each owning 64 batch rows. Hidden states live in LDS in
// CDNA5 WMMA A-operand layout (f16, double-buffered). Weights pre-swizzled to
// f16 WMMA B-operand layout in d_ws. Core op: v_wmma_f32_16x16x32_f16.
// Gate accumulations split into sequential z/r/xh/rh passes; finished gates
// are finalized (bias [+sigmoid]) and packed to f16 immediately. B-tiles are
// software-pipelined one iteration ahead to hide L2 latency. No values are
// kept live across the whole time loop beyond biases/pointers (VGPR budget).
// ---------------------------------------------------------------------------

typedef __attribute__((ext_vector_type(16))) _Float16 v16h;
typedef __attribute__((ext_vector_type(8)))  _Float16 v8h;
typedef __attribute__((ext_vector_type(8)))  float    v8f;

#define UNITS   256
#define NCOLS   768            // 3*UNITS gate columns
#define SEQ     192
#define INDIM   6
#define OUTDIM  5
#define MROWS   64             // batch rows per block (4 WMMA M-tiles)
#define NSTEPS  191            // 64 warm + 127 decode GRU steps

// ws layout (in _Float16 elements):
//  U-style mats (U1=0, W2=1, U2=2): tile(mat,ntile,kt) at
//  (mat*384 + ntile*8 + kt)*512,  ntile in [0,48), kt in [0,8)
#define W1OFF  589824          //  W1 B-tiles (K padded 6->32): ntile*512
#define WDOFF  614400          //  Wd B-tiles: kt*512 (cols 5..15 zero)
#define WSTOT  618496

__device__ __forceinline__ v8f wmma_f16(v16h a, v16h b, v8f c) {
    return __builtin_amdgcn_wmma_f32_16x16x32_f16(
        false, a, false, b, (short)0, c, false, false);
}

__device__ __forceinline__ float sigmf(float x) {
    return 1.0f / (1.0f + __expf(-x));
}

// A-layout tile index: [buf][mt][kt][lane][half]
__device__ __forceinline__ int aidx(int buf, int mt, int kt, int lane) {
    return (((buf * 4 + mt) * 8 + kt) * 32 + lane) * 16;
}

// Element (m, j) inside the A-layout hidden-state buffer.
__device__ __forceinline__ int h_elem_idx(int buf, int m, int j) {
    int kt  = j >> 5;
    int kk  = j & 31;
    int hid = kk & 15;
    int ln  = (m & 15) + ((kk & 16) ? 16 : 0);
    return (((buf * 4 + (m >> 4)) * 8 + kt) * 32 + ln) * 16 + hid;
}

// ---------------------------------------------------------------------------
// Weight prep: f32 -> f16, pre-swizzled to WMMA B-operand layout.
// B-tile element (lane l, half h) = M[kt*32 + (l>=16?16:0) + h][ntile*16 + (l&15)]
// ---------------------------------------------------------------------------
__global__ void gru_prep_weights(const float* __restrict__ W1,
                                 const float* __restrict__ U1,
                                 const float* __restrict__ W2,
                                 const float* __restrict__ U2,
                                 const float* __restrict__ Wd,
                                 _Float16* __restrict__ ws) {
    int idx = blockIdx.x * 256 + threadIdx.x;
    if (idx < W1OFF) {
        int t = idx >> 9, e = idx & 511;
        int l = e >> 4, h = e & 15;
        int mat = t / 384, rem = t - mat * 384;
        int ntile = rem >> 3, kt = rem & 7;
        int k = kt * 32 + ((l >= 16) ? 16 : 0) + h;
        int c = ntile * 16 + (l & 15);
        const float* M = (mat == 0) ? U1 : ((mat == 1) ? W2 : U2);
        ws[idx] = (_Float16)M[k * NCOLS + c];
    } else if (idx < WDOFF) {
        int i = idx - W1OFF;
        int ntile = i >> 9, e = i & 511;
        int l = e >> 4, h = e & 15;
        int k = ((l >= 16) ? 16 : 0) + h;
        int c = ntile * 16 + (l & 15);
        ws[idx] = (k < INDIM) ? (_Float16)W1[k * NCOLS + c] : (_Float16)0.0f;
    } else if (idx < WSTOT) {
        int i = idx - WDOFF;
        int kt = i >> 9, e = i & 511;
        int l = e >> 4, h = e & 15;
        int k = kt * 32 + ((l >= 16) ? 16 : 0) + h;
        int c = l & 15;
        ws[idx] = (c < OUTDIM) ? (_Float16)Wd[k * OUTDIM + c] : (_Float16)0.0f;
    }
}

// ---------------------------------------------------------------------------
// Persistent GRU kernel. One block = 64 batch rows for all 191 steps.
// 512 threads = 16 waves; wave w owns hidden units [16w, 16w+16).
// ---------------------------------------------------------------------------
__global__ void __launch_bounds__(512)
__attribute__((amdgpu_waves_per_eu(4, 4)))
gru_persistent(const float* __restrict__ inp,   // [B][192][6]
               const float* __restrict__ b1,    // [2][768]
               const float* __restrict__ b2,    // [2][768]
               const float* __restrict__ bd,    // [5]
               const _Float16* __restrict__ ws,
               float* __restrict__ out) {       // [B][192][5]
    extern __shared__ char smem[];
    _Float16* h1A = (_Float16*)smem;                 // [2][4][8][32][16]  64KB
    _Float16* h2A = (_Float16*)(smem + 65536);       // [2][4][8][32][16]  64KB
    _Float16* Axs = (_Float16*)(smem + 131072);      // [4][32][16]         4KB
    float*    xl  = (float*)  (smem + 135168);       // [64][6]           1.5KB

    const int tid  = threadIdx.x;
    const int wave = tid >> 5;
    const int lane = tid & 31;
    const int n    = lane & 15;
    const int grp  = lane >> 4;
    const int j0   = wave * 16;            // this wave's hidden-unit base
    const int zt   = wave;                 // gate-column n-tiles
    const int rt   = 16 + wave;
    const int ht   = 32 + wave;
    const int b0   = blockIdx.x * MROWS;
    const int laneoff = lane * 16;

    // per-lane gate biases
    const int cz = j0 + n, cr = 256 + j0 + n, ch = 512 + j0 + n;
    const float zb1  = b1[cz] + b1[NCOLS + cz];
    const float rb1  = b1[cr] + b1[NCOLS + cr];
    const float xhb1 = b1[ch];
    const float rhb1 = b1[NCOLS + ch];
    const float zb2  = b2[cz] + b2[NCOLS + cz];
    const float rb2  = b2[cr] + b2[NCOLS + cr];
    const float xhb2 = b2[ch];
    const float rhb2 = b2[NCOLS + ch];
    const float bdv  = (n < OUTDIM) ? bd[n] : 0.0f;

    // B-tile base pointers for this wave (stride 512 halves per kt)
    const _Float16* U1z = ws + (size_t)(0 * 384 + zt * 8) * 512 + laneoff;
    const _Float16* U1r = ws + (size_t)(0 * 384 + rt * 8) * 512 + laneoff;
    const _Float16* U1h = ws + (size_t)(0 * 384 + ht * 8) * 512 + laneoff;
    const _Float16* W2z = ws + (size_t)(1 * 384 + zt * 8) * 512 + laneoff;
    const _Float16* W2r = ws + (size_t)(1 * 384 + rt * 8) * 512 + laneoff;
    const _Float16* W2h = ws + (size_t)(1 * 384 + ht * 8) * 512 + laneoff;
    const _Float16* U2z = ws + (size_t)(2 * 384 + zt * 8) * 512 + laneoff;
    const _Float16* U2r = ws + (size_t)(2 * 384 + rt * 8) * 512 + laneoff;
    const _Float16* U2h = ws + (size_t)(2 * 384 + ht * 8) * 512 + laneoff;
    const _Float16* W1zp = ws + W1OFF + (size_t)zt * 512 + laneoff;
    const _Float16* W1rp = ws + W1OFF + (size_t)rt * 512 + laneoff;
    const _Float16* W1hp = ws + W1OFF + (size_t)ht * 512 + laneoff;
    const _Float16* WdB  = ws + WDOFF + laneoff;

    // init: h1 = h2 = 0 in buffer 0, zero Ax pad region, load step-0 inputs
    for (int i = tid; i < 16384; i += 512) {
        h1A[i] = (_Float16)0.0f;
        h2A[i] = (_Float16)0.0f;
    }
    for (int i = tid; i < 2048; i += 512) Axs[i] = (_Float16)0.0f;
    if (tid < MROWS * INDIM) {
        int m = tid / INDIM, k = tid - m * INDIM;
        xl[m * INDIM + k] = inp[((size_t)(b0 + m) * SEQ + 0) * INDIM + k];
    }
    __syncthreads();

    const v8f vzero = {};

    for (int s = 0; s < NSTEPS; ++s) {
        const int cur = s & 1;
        const int nxt = cur ^ 1;

        // ---- build x A-tile (K=0..5 valid, rest stays zero) ----
        if (tid < MROWS * INDIM) {
            int m = tid / INDIM, k = tid - m * INDIM;
            Axs[((m >> 4) * 32 + (m & 15)) * 16 + k] = (_Float16)xl[m * INDIM + k];
        }
        __syncthreads();

        // =================== LAYER 1 ===================
        v8h zpk[4], rpk[4], xhpk[4];   // finalized gates, packed f16
        v8f acc[4];

        // z = sigmoid(x@W1z + h1@U1z + zb1)
        {
            v16h bw = *(const v16h*)W1zp;
            #pragma unroll
            for (int mt = 0; mt < 4; ++mt) {
                v16h a = *(const v16h*)(Axs + (mt * 32 + lane) * 16);
                acc[mt] = wmma_f16(a, bw, vzero);
            }
        }
        {
            v16h b = *(const v16h*)U1z;
            #pragma unroll 1
            for (int kt = 0; kt < 8; ++kt) {
                int ktn = (kt < 7) ? kt + 1 : 7;
                v16h bn = *(const v16h*)(U1z + (size_t)ktn * 512);
                #pragma unroll
                for (int mt = 0; mt < 4; ++mt) {
                    v16h a = *(const v16h*)(h1A + aidx(cur, mt, kt, lane));
                    acc[mt] = wmma_f16(a, b, acc[mt]);
                }
                b = bn;
            }
        }
        #pragma unroll
        for (int mt = 0; mt < 4; ++mt)
            #pragma unroll
            for (int e = 0; e < 8; ++e)
                zpk[mt][e] = (_Float16)sigmf(acc[mt][e] + zb1);

        // r = sigmoid(x@W1r + h1@U1r + rb1)
        {
            v16h bw = *(const v16h*)W1rp;
            #pragma unroll
            for (int mt = 0; mt < 4; ++mt) {
                v16h a = *(const v16h*)(Axs + (mt * 32 + lane) * 16);
                acc[mt] = wmma_f16(a, bw, vzero);
            }
        }
        {
            v16h b = *(const v16h*)U1r;
            #pragma unroll 1
            for (int kt = 0; kt < 8; ++kt) {
                int ktn = (kt < 7) ? kt + 1 : 7;
                v16h bn = *(const v16h*)(U1r + (size_t)ktn * 512);
                #pragma unroll
                for (int mt = 0; mt < 4; ++mt) {
                    v16h a = *(const v16h*)(h1A + aidx(cur, mt, kt, lane));
                    acc[mt] = wmma_f16(a, b, acc[mt]);
                }
                b = bn;
            }
        }
        #pragma unroll
        for (int mt = 0; mt < 4; ++mt)
            #pragma unroll
            for (int e = 0; e < 8; ++e)
                rpk[mt][e] = (_Float16)sigmf(acc[mt][e] + rb1);

        // xh = x@W1h + xhb1 (kept separate: r gates only the recurrent part)
        {
            v16h bw = *(const v16h*)W1hp;
            #pragma unroll
            for (int mt = 0; mt < 4; ++mt) {
                v16h a = *(const v16h*)(Axs + (mt * 32 + lane) * 16);
                v8f t = wmma_f16(a, bw, vzero);
                #pragma unroll
                for (int e = 0; e < 8; ++e)
                    xhpk[mt][e] = (_Float16)(t[e] + xhb1);
            }
        }

        // rh = h1@U1h  (stays f32 until combine)
        #pragma unroll
        for (int i = 0; i < 4; ++i) acc[i] = vzero;
        {
            v16h b = *(const v16h*)U1h;
            #pragma unroll 1
            for (int kt = 0; kt < 8; ++kt) {
                int ktn = (kt < 7) ? kt + 1 : 7;
                v16h bn = *(const v16h*)(U1h + (size_t)ktn * 512);
                #pragma unroll
                for (int mt = 0; mt < 4; ++mt) {
                    v16h a = *(const v16h*)(h1A + aidx(cur, mt, kt, lane));
                    acc[mt] = wmma_f16(a, b, acc[mt]);
                }
                b = bn;
            }
        }
        // combine -> h1_new (write to the other buffer)
        #pragma unroll
        for (int mt = 0; mt < 4; ++mt) {
            #pragma unroll
            for (int e = 0; e < 8; ++e) {
                int m = mt * 16 + grp * 8 + e;
                float z  = (float)zpk[mt][e];
                float r  = (float)rpk[mt][e];
                float hh = tanhf((float)xhpk[mt][e] + r * (acc[mt][e] + rhb1));
                int j = j0 + n;
                float hold = (float)h1A[h_elem_idx(cur, m, j)];
                float hnew = z * hold + (1.0f - z) * hh;
                h1A[h_elem_idx(nxt, m, j)] = (_Float16)hnew;
            }
        }
        __syncthreads();

        // =================== LAYER 2 ===================
        // z = sigmoid(h1_new@W2z + h2@U2z + zb2)
        #pragma unroll
        for (int i = 0; i < 4; ++i) acc[i] = vzero;
        {
            v16h bx = *(const v16h*)W2z;
            v16h bh = *(const v16h*)U2z;
            #pragma unroll 1
            for (int kt = 0; kt < 8; ++kt) {
                int ktn = (kt < 7) ? kt + 1 : 7;
                v16h bxn = *(const v16h*)(W2z + (size_t)ktn * 512);
                v16h bhn = *(const v16h*)(U2z + (size_t)ktn * 512);
                #pragma unroll
                for (int mt = 0; mt < 4; ++mt) {
                    v16h a1 = *(const v16h*)(h1A + aidx(nxt, mt, kt, lane));
                    v16h a2 = *(const v16h*)(h2A + aidx(cur, mt, kt, lane));
                    acc[mt] = wmma_f16(a1, bx, acc[mt]);
                    acc[mt] = wmma_f16(a2, bh, acc[mt]);
                }
                bx = bxn;
                bh = bhn;
            }
        }
        #pragma unroll
        for (int mt = 0; mt < 4; ++mt)
            #pragma unroll
            for (int e = 0; e < 8; ++e)
                zpk[mt][e] = (_Float16)sigmf(acc[mt][e] + zb2);

        // r = sigmoid(h1_new@W2r + h2@U2r + rb2)
        #pragma unroll
        for (int i = 0; i < 4; ++i) acc[i] = vzero;
        {
            v16h bx = *(const v16h*)W2r;
            v16h bh = *(const v16h*)U2r;
            #pragma unroll 1
            for (int kt = 0; kt < 8; ++kt) {
                int ktn = (kt < 7) ? kt + 1 : 7;
                v16h bxn = *(const v16h*)(W2r + (size_t)ktn * 512);
                v16h bhn = *(const v16h*)(U2r + (size_t)ktn * 512);
                #pragma unroll
                for (int mt = 0; mt < 4; ++mt) {
                    v16h a1 = *(const v16h*)(h1A + aidx(nxt, mt, kt, lane));
                    v16h a2 = *(const v16h*)(h2A + aidx(cur, mt, kt, lane));
                    acc[mt] = wmma_f16(a1, bx, acc[mt]);
                    acc[mt] = wmma_f16(a2, bh, acc[mt]);
                }
                bx = bxn;
                bh = bhn;
            }
        }
        #pragma unroll
        for (int mt = 0; mt < 4; ++mt)
            #pragma unroll
            for (int e = 0; e < 8; ++e)
                rpk[mt][e] = (_Float16)sigmf(acc[mt][e] + rb2);

        // xh = h1_new@W2h + xhb2
        #pragma unroll
        for (int i = 0; i < 4; ++i) acc[i] = vzero;
        {
            v16h b = *(const v16h*)W2h;
            #pragma unroll 1
            for (int kt = 0; kt < 8; ++kt) {
                int ktn = (kt < 7) ? kt + 1 : 7;
                v16h bn = *(const v16h*)(W2h + (size_t)ktn * 512);
                #pragma unroll
                for (int mt = 0; mt < 4; ++mt) {
                    v16h a = *(const v16h*)(h1A + aidx(nxt, mt, kt, lane));
                    acc[mt] = wmma_f16(a, b, acc[mt]);
                }
                b = bn;
            }
        }
        #pragma unroll
        for (int mt = 0; mt < 4; ++mt)
            #pragma unroll
            for (int e = 0; e < 8; ++e)
                xhpk[mt][e] = (_Float16)(acc[mt][e] + xhb2);

        // rh = h2@U2h  (stays f32 until combine)
        #pragma unroll
        for (int i = 0; i < 4; ++i) acc[i] = vzero;
        {
            v16h b = *(const v16h*)U2h;
            #pragma unroll 1
            for (int kt = 0; kt < 8; ++kt) {
                int ktn = (kt < 7) ? kt + 1 : 7;
                v16h bn = *(const v16h*)(U2h + (size_t)ktn * 512);
                #pragma unroll
                for (int mt = 0; mt < 4; ++mt) {
                    v16h a = *(const v16h*)(h2A + aidx(cur, mt, kt, lane));
                    acc[mt] = wmma_f16(a, b, acc[mt]);
                }
                b = bn;
            }
        }
        // combine -> h2_new
        #pragma unroll
        for (int mt = 0; mt < 4; ++mt) {
            #pragma unroll
            for (int e = 0; e < 8; ++e) {
                int m = mt * 16 + grp * 8 + e;
                float z  = (float)zpk[mt][e];
                float r  = (float)rpk[mt][e];
                float hh = tanhf((float)xhpk[mt][e] + r * (acc[mt][e] + rhb2));
                int j = j0 + n;
                float hold = (float)h2A[h_elem_idx(cur, m, j)];
                float hnew = z * hold + (1.0f - z) * hh;
                h2A[h_elem_idx(nxt, m, j)] = (_Float16)hnew;
            }
        }
        __syncthreads();

        // =================== OUTPUT / NEXT INPUT ===================
        if (wave < 4) {
            const int mt = wave;
            v8f pacc = vzero;
            v16h b = *(const v16h*)WdB;
            #pragma unroll 1
            for (int kt = 0; kt < 8; ++kt) {
                int ktn = (kt < 7) ? kt + 1 : 7;
                v16h bn = *(const v16h*)(WdB + (size_t)ktn * 512);
                v16h a = *(const v16h*)(h2A + aidx(nxt, mt, kt, lane));
                pacc = wmma_f16(a, b, pacc);
                b = bn;
            }
            #pragma unroll
            for (int e = 0; e < 8; ++e) {
                int m = mt * 16 + grp * 8 + e;
                float p = pacc[e] + bdv;
                if (n < OUTDIM) {
                    int tout = (s < 64) ? s : (s + 1);
                    out[((size_t)(b0 + m) * SEQ + tout) * OUTDIM + n] = p;
                    if (s == 63)
                        out[((size_t)(b0 + m) * SEQ + 64) * OUTDIM + n] = p;
                    if (s >= 63)
                        xl[m * INDIM + 1 + n] = p;   // feedback for decode
                }
            }
        }
        // stage next-step exogenous input
        if (s + 1 < 64) {
            if (tid < MROWS * INDIM) {
                int m = tid / INDIM, k = tid - m * INDIM;
                xl[m * INDIM + k] = inp[((size_t)(b0 + m) * SEQ + (s + 1)) * INDIM + k];
            }
        } else {
            if (tid < MROWS) {
                xl[tid * INDIM] = inp[((size_t)(b0 + tid) * SEQ + (s + 1)) * INDIM];
            }
        }
        __syncthreads();
    }
}

// ---------------------------------------------------------------------------
extern "C" void kernel_launch(void* const* d_in, const int* in_sizes, int n_in,
                              void* d_out, int out_size, void* d_ws, size_t ws_size,
                              hipStream_t stream) {
    const float* inp = (const float*)d_in[0];
    const float* W1  = (const float*)d_in[1];
    const float* U1  = (const float*)d_in[2];
    const float* b1  = (const float*)d_in[3];
    const float* W2  = (const float*)d_in[4];
    const float* U2  = (const float*)d_in[5];
    const float* b2  = (const float*)d_in[6];
    const float* Wd  = (const float*)d_in[7];
    const float* bd  = (const float*)d_in[8];
    _Float16* ws = (_Float16*)d_ws;
    float* out = (float*)d_out;

    gru_prep_weights<<<WSTOT / 256, 256, 0, stream>>>(W1, U1, W2, U2, Wd, ws);

    int Btot = in_sizes[0] / (SEQ * INDIM);
    int nblk = Btot / MROWS;
    size_t shmem = 65536 + 65536 + 4096 + 1536;
    gru_persistent<<<nblk, 512, shmem, stream>>>(inp, b1, b2, bd, ws, out);

    (void)n_in; (void)out_size; (void)ws_size;
}
